// Transformer_455266533409
// MI455X (gfx1250) — matrix operand order repaired and checked by reading, not compiled, not run on hardware
//
#include <hip/hip_runtime.h>

#define Bc 16
#define Tc 512
#define Vc 2048
#define Dc 512
#define HDc 64
#define NHc 8
#define Lc 6
#define Fc 2048
#define MDc 16
#define Mrows (Bc*Tc)          // 8192
#define NJ (2*MDc+1)           // 33
#define KC 64                  // K-chunk staged in LDS
#define BPAD 72                // halfs per LDS B row (64 + 8 pad, 144B -> spreads banks)

typedef __attribute__((ext_vector_type(16))) _Float16 v16h;
typedef __attribute__((ext_vector_type(8)))  _Float16 v8h;
typedef __attribute__((ext_vector_type(8)))  float    v8f;

union Frag { v16h v; v8h h2[2]; };

__device__ inline v8f wmma16(v16h a, v16h b, v8f c) {
  return __builtin_amdgcn_wmma_f32_16x16x32_f16(false, a, false, b, (short)0, c, false, false);
}

// Async DMA: global (saddr64 + vaddr32) -> LDS[lds_off], 16 bytes per lane. ASYNCcnt-tracked.
__device__ inline void async_b128(unsigned lds_off, unsigned gvoff, const void* sbase) {
  asm volatile("global_load_async_to_lds_b128 %0, %1, %2"
               :: "v"(lds_off), "v"(gvoff), "s"(sbase) : "memory");
}
__device__ inline void wait_async0() {
  asm volatile("s_wait_asynccnt 0x0" ::: "memory");
}

// ---------------- embedding + positional encoding: h = 2*emb[idx] + pe ----------------
__global__ void embed_kernel(const int* __restrict__ idx, const float* __restrict__ emb,
                             float* __restrict__ h) {
  int i = blockIdx.x * 256 + threadIdx.x;
  if (i >= Mrows * Dc) return;
  int d  = i % Dc;
  int bt = i / Dc;
  int t  = bt % Tc;
  int tok = idx[bt];
  float ie = (float)(d & ~1);   // replicate reference "bug": exponent 2*i/D with i already even
  float ang = (float)t * __powf(10000.0f, -2.0f * ie / (float)Dc);
  float pe = (d & 1) ? __cosf(ang) : __sinf(ang);
  h[i] = 2.0f * emb[(size_t)tok * Dc + d] + pe;
}

// ---------------- weight conversion: f32 [K][N] row-major -> f16 [N][K] (transposed) ----------------
__global__ void convT_kernel(const float* __restrict__ in, _Float16* __restrict__ out,
                             int Kdim, int Ndim) {
  int i = blockIdx.x * 256 + threadIdx.x;
  if (i >= Kdim * Ndim) return;
  int n = i % Ndim;
  int k = i / Ndim;
  out[(size_t)n * Kdim + k] = (_Float16)in[i];
}

// Wsum[h][d] = sum_n Wo[n*HD+h][d]; stored transposed as out[d][h] (f16)
__global__ void wsum_kernel(const float* __restrict__ Wo_l, _Float16* __restrict__ out) {
  int i = blockIdx.x * 256 + threadIdx.x;
  if (i >= Dc * HDc) return;
  int hh = i % HDc;
  int d  = i / HDc;
  float s = 0.f;
#pragma unroll
  for (int n = 0; n < NHc; ++n) s += Wo_l[(size_t)(n * HDc + hh) * Dc + d];
  out[(size_t)d * HDc + hh] = (_Float16)s;
}

__global__ void cvt16_kernel(const float* __restrict__ in, _Float16* __restrict__ out, int n) {
  int i = blockIdx.x * 256 + threadIdx.x;
  if (i < n) out[i] = (_Float16)in[i];
}

// ---------------- LayerNorm row kernel: y_f16 = LN(h)*g + b ----------------
__global__ __launch_bounds__(256)
void ln_kernel(const float* __restrict__ h, const float* __restrict__ g,
               const float* __restrict__ be, _Float16* __restrict__ y) {
  int row = blockIdx.x;
  const float* x = h + (size_t)row * Dc;
  float s = 0.f, s2 = 0.f;
  for (int d = threadIdx.x; d < Dc; d += 256) { float v = x[d]; s += v; s2 += v * v; }
#pragma unroll
  for (int off = 16; off; off >>= 1) { s += __shfl_xor(s, off); s2 += __shfl_xor(s2, off); }
  __shared__ float red0[8], red1[8];
  int wave = threadIdx.x >> 5, lid = threadIdx.x & 31;
  if (lid == 0) { red0[wave] = s; red1[wave] = s2; }
  __syncthreads();
  if (threadIdx.x == 0) {
    float a = 0.f, b2 = 0.f;
#pragma unroll
    for (int i = 0; i < 8; ++i) { a += red0[i]; b2 += red1[i]; }
    red0[0] = a; red1[0] = b2;
  }
  __syncthreads();
  float mean = red0[0] / (float)Dc;
  float var  = red1[0] / (float)Dc - mean * mean;
  float inv  = rsqrtf(var + 1e-5f);
  for (int d = threadIdx.x; d < Dc; d += 256)
    y[(size_t)row * Dc + d] = (_Float16)((x[d] - mean) * inv * g[d] + be[d]);
}

// ---------------- WMMA GEMM with async-LDS staged B ----------------
// A: f16 [M][K] row-major.  Bt: f16 [N][K] (B transposed, K contiguous).
// Block = 256 threads = 8 waves; block tile = 128(M) x 64(N); wave tile = 16 x 64.
// B 64x64 k-chunks are DMA'd to LDS with global_load_async_to_lds_b128, double-buffered.
// Inner loop preloads ALL fragments of a chunk (2 A-frags + 8 B-frags), then bursts 8 WMMAs.
// Outputs (each optional): Cf (f32 [M][N]), Ch (f16 [M][N]), Cht (f16 [N][M]).
__global__ __launch_bounds__(256)
void gemm_wmma(const _Float16* __restrict__ A, const _Float16* __restrict__ Bt,
               const float* __restrict__ bias, const float* __restrict__ res,
               float* __restrict__ Cf, _Float16* __restrict__ Ch, _Float16* __restrict__ Cht,
               int Mdim, int Ndim, int Kdim, int relu) {
  __shared__ _Float16 ldsb[2][64 * BPAD];
  const int tid  = threadIdx.x;
  const int wave = tid >> 5;
  const int lid  = tid & 31;
  const int m0   = blockIdx.x * 128 + wave * 16;
  const int n0   = blockIdx.y * 64;
  const int mrow = lid & 15, sel = lid >> 4;

  // Staging map: 512 x b128 segments per 64x64 chunk; 2 per thread.
  const int sA = tid * 2, sB = tid * 2 + 1;
  const int rA = sA >> 3, cA = sA & 7;
  const int rB = sB >> 3, cB = sB & 7;
  const unsigned goffA = (unsigned)((((size_t)(n0 + rA)) * Kdim + cA * 8) * 2);
  const unsigned goffB = (unsigned)((((size_t)(n0 + rB)) * Kdim + cB * 8) * 2);
  const unsigned ldsbase = (unsigned)(unsigned long long)(uintptr_t)&ldsb[0][0];
  const unsigned loffA = (unsigned)((rA * BPAD + cA * 8) * 2);
  const unsigned loffB = (unsigned)((rB * BPAD + cB * 8) * 2);
  const unsigned bufstride = 64 * BPAD * 2;

  v8f acc0 = {}, acc1 = {}, acc2 = {}, acc3 = {};
  const _Float16* Arow = A + (size_t)(m0 + mrow) * Kdim;

  const int nch = Kdim / KC;
  async_b128(ldsbase + loffA, goffA, Bt);
  async_b128(ldsbase + loffB, goffB, Bt);
  wait_async0();
  __syncthreads();

  for (int c = 0; c < nch; ++c) {
    if (c + 1 < nch) {
      unsigned base = ldsbase + ((c + 1) & 1) * bufstride;
      unsigned kb2 = (unsigned)((c + 1) * KC * 2);
      async_b128(base + loffA, goffA + kb2, Bt);
      async_b128(base + loffB, goffB + kb2, Bt);
    }
    const _Float16* ldsB = &ldsb[c & 1][0];
    const int kb = c * KC;

    // ---- preload whole chunk into registers ----
    Frag a[2];
#pragma unroll
    for (int ks = 0; ks < 2; ++ks) {
      a[ks].h2[0] = *(const v8h*)(Arow + kb + ks * 32 + sel * 8);
      a[ks].h2[1] = *(const v8h*)(Arow + kb + ks * 32 + 16 + sel * 8);
    }
    Frag b[2][4];
#pragma unroll
    for (int ks = 0; ks < 2; ++ks)
#pragma unroll
      for (int nt = 0; nt < 4; ++nt) {
        const _Float16* Bp = ldsB + (nt * 16 + mrow) * BPAD + ks * 32;
        b[ks][nt].h2[0] = *(const v8h*)(Bp + sel * 8);
        b[ks][nt].h2[1] = *(const v8h*)(Bp + 16 + sel * 8);
      }

    // ---- 8 WMMAs back-to-back ----
#pragma unroll
    for (int ks = 0; ks < 2; ++ks) {
      acc0 = wmma16(a[ks].v, b[ks][0].v, acc0);
      acc1 = wmma16(a[ks].v, b[ks][1].v, acc1);
      acc2 = wmma16(a[ks].v, b[ks][2].v, acc2);
      acc3 = wmma16(a[ks].v, b[ks][3].v, acc3);
    }

    wait_async0();      // next chunk's DMA into LDS complete
    __syncthreads();    // all waves done reading current buf; staged buf visible
  }

#pragma unroll
  for (int nt = 0; nt < 4; ++nt) {
    v8f acc = nt == 0 ? acc0 : nt == 1 ? acc1 : nt == 2 ? acc2 : acc3;
    int n = n0 + nt * 16 + mrow;
    float bvs = bias ? bias[n] : 0.f;
#pragma unroll
    for (int r = 0; r < 8; ++r) {
      int m = m0 + sel * 8 + r;
      float v = acc[r] + bvs;
      if (relu) v = fmaxf(v, 0.f);
      if (res) v += res[(size_t)m * Ndim + n];
      if (Cf)  Cf[(size_t)m * Ndim + n] = v;
      if (Ch)  Ch[(size_t)m * Ndim + n] = (_Float16)v;
      if (Cht) Cht[(size_t)n * Mdim + m] = (_Float16)v;
    }
  }
}

// ---------------- P = Q @ rel_table^T : [M][33] ----------------
__global__ void p_kernel(const _Float16* __restrict__ Qh, const float* __restrict__ rel,
                         float* __restrict__ P) {
  int i = blockIdx.x * 256 + threadIdx.x;
  if (i >= Mrows * NJ) return;
  int j  = i % NJ;
  int bt = i / NJ;
  float s = 0.f;
#pragma unroll
  for (int hh = 0; hh < HDc; ++hh)
    s += (float)Qh[(size_t)bt * HDc + hh] * rel[(size_t)j * HDc + hh];
  P[i] = s;
}

// ---------------- fused attention: scores(WMMA) -> softmax -> att@V(WMMA) + A2@rel ----------------
__global__ __launch_bounds__(32)
void attn_kernel(const _Float16* __restrict__ Qh, const _Float16* __restrict__ Kh,
                 const _Float16* __restrict__ Vt, const float* __restrict__ P,
                 const float* __restrict__ rel, _Float16* __restrict__ valsH) {
  __shared__ float    s_sc[16][Tc];   // 32 KB scores
  __shared__ _Float16 s_at[16][Tc];   // 16 KB att (f16)
  __shared__ float    s_a2[16][17];   // bucketed attention mass
  int b  = blockIdx.x >> 5;           // Tc/16 = 32 tiles
  int t0 = (blockIdx.x & 31) * 16;
  int lid = threadIdx.x;
  int mrow = lid & 15, sel = lid >> 4;
  const float scale = 0.125f;         // 1/sqrt(64)

  Frag aQ0, aQ1;
  const _Float16* Qrow = Qh + (size_t)(b * Tc + t0 + mrow) * HDc;
  aQ0.h2[0] = *(const v8h*)(Qrow + sel * 8);
  aQ0.h2[1] = *(const v8h*)(Qrow + 16 + sel * 8);
  aQ1.h2[0] = *(const v8h*)(Qrow + 32 + sel * 8);
  aQ1.h2[1] = *(const v8h*)(Qrow + 48 + sel * 8);

  for (int st = 0; st < Tc / 16; ++st) {
    int s0 = st * 16;
    v8f acc = {};
#pragma unroll
    for (int ks = 0; ks < 2; ++ks) {
      Frag bf;
      const _Float16* Krow = Kh + (size_t)(b * Tc + s0 + mrow) * HDc + ks * 32;
      bf.h2[0] = *(const v8h*)(Krow + sel * 8);
      bf.h2[1] = *(const v8h*)(Krow + 16 + sel * 8);
      acc = wmma16(ks ? aQ1.v : aQ0.v, bf.v, acc);
    }
    int s = s0 + mrow;
#pragma unroll
    for (int r = 0; r < 8; ++r) {
      int mr = sel * 8 + r;
      int t = t0 + mr;
      int dj = s - t; dj = dj < -MDc ? -MDc : (dj > MDc ? MDc : dj);
      float v = (acc[r] + P[(size_t)(b * Tc + t) * NJ + dj + MDc]) * scale;
      if (s > t) v = -1e30f;
      s_sc[mr][s] = v;
    }
  }
  __syncthreads();

  for (int row = 0; row < 16; ++row) {
    int t = t0 + row;
    float mx = -1e30f;
    for (int s = lid; s < Tc; s += 32) mx = fmaxf(mx, s_sc[row][s]);
#pragma unroll
    for (int off = 16; off; off >>= 1) mx = fmaxf(mx, __shfl_xor(mx, off));
    float sum = 0.f;
    for (int s = lid; s < Tc; s += 32) { float e = __expf(s_sc[row][s] - mx); s_sc[row][s] = e; sum += e; }
#pragma unroll
    for (int off = 16; off; off >>= 1) sum += __shfl_xor(sum, off);
    float inv = 1.0f / sum;
    float j0 = 0.f;
    for (int s = lid; s < Tc; s += 32) {
      float a = s_sc[row][s] * inv;
      s_at[row][s] = (_Float16)a;
      if (s <= t - MDc) j0 += a;
    }
#pragma unroll
    for (int off = 16; off; off >>= 1) j0 += __shfl_xor(j0, off);
    if (lid == 0) s_a2[row][0] = j0;
    if (lid >= 1 && lid <= 16) {
      int s = t + lid - MDc;
      s_a2[row][lid] = (s >= 0) ? s_sc[row][s] * inv : 0.f;
    }
  }
  __syncthreads();

  v8f accV0 = {}, accV1 = {}, accV2 = {}, accV3 = {};
  for (int st = 0; st < Tc / 32; ++st) {
    Frag aA;
    aA.h2[0] = *(const v8h*)(&s_at[mrow][st * 32 + sel * 8]);
    aA.h2[1] = *(const v8h*)(&s_at[mrow][st * 32 + 16 + sel * 8]);
    Frag bf0, bf1, bf2, bf3;
#pragma unroll
    for (int nt = 0; nt < 4; ++nt) {
      const _Float16* Vrow = Vt + (size_t)(nt * 16 + mrow) * Mrows + b * Tc + st * 32;
      Frag* bp = nt == 0 ? &bf0 : nt == 1 ? &bf1 : nt == 2 ? &bf2 : &bf3;
      bp->h2[0] = *(const v8h*)(Vrow + sel * 8);
      bp->h2[1] = *(const v8h*)(Vrow + 16 + sel * 8);
    }
    accV0 = wmma16(aA.v, bf0.v, accV0);
    accV1 = wmma16(aA.v, bf1.v, accV1);
    accV2 = wmma16(aA.v, bf2.v, accV2);
    accV3 = wmma16(aA.v, bf3.v, accV3);
  }
#pragma unroll
  for (int nt = 0; nt < 4; ++nt) {
    v8f acc = nt == 0 ? accV0 : nt == 1 ? accV1 : nt == 2 ? accV2 : accV3;
    int hh = nt * 16 + mrow;
#pragma unroll
    for (int r = 0; r < 8; ++r) {
      int mr = sel * 8 + r;
      float rv = 0.f;
#pragma unroll
      for (int j = 0; j <= 16; ++j) rv += s_a2[mr][j] * rel[(size_t)j * HDc + hh];
      valsH[(size_t)(b * Tc + t0 + mr) * HDc + hh] = (_Float16)(acc[r] + rv);
    }
  }
}

// ---------------- host side ----------------
extern "C" void kernel_launch(void* const* d_in, const int* in_sizes, int n_in,
                              void* d_out, int out_size, void* d_ws, size_t ws_size,
                              hipStream_t stream) {
  (void)in_sizes; (void)n_in; (void)out_size; (void)ws_size;
  const int*   idx  = (const int*)d_in[0];
  const float* emb  = (const float*)d_in[1];
  const float* Wq   = (const float*)d_in[2];
  const float* bq   = (const float*)d_in[3];
  const float* Wk   = (const float*)d_in[4];
  const float* bk   = (const float*)d_in[5];
  const float* Wv   = (const float*)d_in[6];
  const float* bv   = (const float*)d_in[7];
  const float* rel  = (const float*)d_in[8];
  const float* Wo   = (const float*)d_in[9];
  const float* bo   = (const float*)d_in[10];
  const float* g1   = (const float*)d_in[11];
  const float* be1  = (const float*)d_in[12];
  const float* g2   = (const float*)d_in[13];
  const float* be2  = (const float*)d_in[14];
  const float* W1   = (const float*)d_in[15];
  const float* b1   = (const float*)d_in[16];
  const float* W2   = (const float*)d_in[17];
  const float* b2   = (const float*)d_in[18];
  const float* Wout = (const float*)d_in[19];
  const float* bout = (const float*)d_in[20];

  char* p = (char*)d_ws;
  auto carve = [&](size_t bytes) -> char* {
    char* r = p; p += (bytes + 255) & ~(size_t)255; return r;
  };
  float*    h      = (float*)   carve((size_t)Mrows * Dc * 4);
  _Float16* y16    = (_Float16*)carve((size_t)Mrows * Dc * 2);
  _Float16* q16    = (_Float16*)carve((size_t)Mrows * HDc * 2);
  _Float16* k16    = (_Float16*)carve((size_t)Mrows * HDc * 2);
  _Float16* vt16   = (_Float16*)carve((size_t)HDc * Mrows * 2);
  _Float16* vals16 = (_Float16*)carve((size_t)Mrows * HDc * 2);
  float*    P      = (float*)   carve((size_t)Mrows * NJ * 4);
  _Float16* z16    = (_Float16*)carve((size_t)Mrows * Fc * 2);
  _Float16* wqt    = (_Float16*)carve((size_t)Lc * HDc * Dc * 2);
  _Float16* wkt    = (_Float16*)carve((size_t)Lc * HDc * Dc * 2);
  _Float16* wvt    = (_Float16*)carve((size_t)Lc * HDc * Dc * 2);
  _Float16* wsum   = (_Float16*)carve((size_t)Lc * Dc * HDc * 2);
  _Float16* w1t    = (_Float16*)carve((size_t)Lc * Fc * Dc * 2);
  _Float16* w2t    = (_Float16*)carve((size_t)Lc * Dc * Fc * 2);
  _Float16* woutt  = (_Float16*)carve((size_t)Vc * Dc * 2);

  embed_kernel<<<(Mrows * Dc + 255) / 256, 256, 0, stream>>>(idx, emb, h);
  for (int l = 0; l < Lc; ++l) {
    convT_kernel<<<(Dc * HDc + 255) / 256, 256, 0, stream>>>(Wq + (size_t)l * Dc * HDc, wqt + (size_t)l * HDc * Dc, Dc, HDc);
    convT_kernel<<<(Dc * HDc + 255) / 256, 256, 0, stream>>>(Wk + (size_t)l * Dc * HDc, wkt + (size_t)l * HDc * Dc, Dc, HDc);
    convT_kernel<<<(Dc * HDc + 255) / 256, 256, 0, stream>>>(Wv + (size_t)l * Dc * HDc, wvt + (size_t)l * HDc * Dc, Dc, HDc);
    wsum_kernel<<<(Dc * HDc + 255) / 256, 256, 0, stream>>>(Wo + (size_t)l * Dc * Dc, wsum + (size_t)l * Dc * HDc);
    convT_kernel<<<(Dc * Fc + 255) / 256, 256, 0, stream>>>(W1 + (size_t)l * Dc * Fc, w1t + (size_t)l * Fc * Dc, Dc, Fc);
    convT_kernel<<<(Fc * Dc + 255) / 256, 256, 0, stream>>>(W2 + (size_t)l * Fc * Dc, w2t + (size_t)l * Dc * Fc, Fc, Dc);
  }
  convT_kernel<<<(Dc * Vc + 255) / 256, 256, 0, stream>>>(Wout, woutt, Dc, Vc);

  for (int l = 0; l < Lc; ++l) {
    const _Float16* wqt_l  = wqt  + (size_t)l * HDc * Dc;
    const _Float16* wkt_l  = wkt  + (size_t)l * HDc * Dc;
    const _Float16* wvt_l  = wvt  + (size_t)l * HDc * Dc;
    const _Float16* wsum_l = wsum + (size_t)l * Dc * HDc;
    const _Float16* w1t_l  = w1t  + (size_t)l * Fc * Dc;
    const _Float16* w2t_l  = w2t  + (size_t)l * Dc * Fc;
    const float*    rel_l  = rel  + (size_t)l * NJ * HDc;

    ln_kernel<<<Mrows, 256, 0, stream>>>(h, g1 + (size_t)l * Dc, be1 + (size_t)l * Dc, y16);
    gemm_wmma<<<dim3(Mrows / 128, HDc / 64), 256, 0, stream>>>(y16, wqt_l, bq + (size_t)l * HDc,
        nullptr, nullptr, q16, nullptr, Mrows, HDc, Dc, 0);
    gemm_wmma<<<dim3(Mrows / 128, HDc / 64), 256, 0, stream>>>(y16, wkt_l, bk + (size_t)l * HDc,
        nullptr, nullptr, k16, nullptr, Mrows, HDc, Dc, 0);
    gemm_wmma<<<dim3(Mrows / 128, HDc / 64), 256, 0, stream>>>(y16, wvt_l, bv + (size_t)l * HDc,
        nullptr, nullptr, nullptr, vt16, Mrows, HDc, Dc, 0);
    p_kernel<<<(Mrows * NJ + 255) / 256, 256, 0, stream>>>(q16, rel_l, P);
    attn_kernel<<<Bc * (Tc / 16), 32, 0, stream>>>(q16, k16, vt16, P, rel_l, vals16);
    gemm_wmma<<<dim3(Mrows / 128, Dc / 64), 256, 0, stream>>>(vals16, wsum_l, bo + (size_t)l * Dc,
        h, h, nullptr, nullptr, Mrows, Dc, HDc, 0);          // h += vals@Wsum + bo
    ln_kernel<<<Mrows, 256, 0, stream>>>(h, g2 + (size_t)l * Dc, be2 + (size_t)l * Dc, y16);
    gemm_wmma<<<dim3(Mrows / 128, Fc / 64), 256, 0, stream>>>(y16, w1t_l, b1 + (size_t)l * Fc,
        nullptr, nullptr, z16, nullptr, Mrows, Fc, Dc, 1);   // z = relu(y@W1 + b1), f16
    gemm_wmma<<<dim3(Mrows / 128, Dc / 64), 256, 0, stream>>>(z16, w2t_l, b2 + (size_t)l * Dc,
        h, h, nullptr, nullptr, Mrows, Dc, Fc, 0);           // h += z@W2 + b2
  }

  cvt16_kernel<<<(Mrows * Dc + 255) / 256, 256, 0, stream>>>(h, y16, Mrows * Dc);
  gemm_wmma<<<dim3(Mrows / 128, Vc / 64), 256, 0, stream>>>(y16, woutt, bout,
      nullptr, (float*)d_out, nullptr, nullptr, Mrows, Vc, Dc, 0);
}
